// TGNMemory_14181982011697
// MI455X (gfx1250) — matrix operand order repaired and checked by reading, not compiled, not run on hardware
//
#include <hip/hip_runtime.h>
#include <hip/hip_bf16.h>

#define DIM   128
#define MT    16          // nodes per workgroup (M tile)
#define GDIM  384         // 3 * DIM gate outputs

typedef __attribute__((ext_vector_type(16))) __bf16 v16bf;
typedef __attribute__((ext_vector_type(8)))  __bf16 v8bf;
typedef __attribute__((ext_vector_type(8)))  float  v8f;

// ---------------------------------------------------------------------------
// Kernel 1: zero agg-sum and counts in workspace
// ---------------------------------------------------------------------------
__global__ __launch_bounds__(256) void tgn_zero_kernel(float* __restrict__ buf,
                                                       int total) {
    int i = blockIdx.x * 256 + threadIdx.x;
    if (i < total) buf[i] = 0.0f;
}

// ---------------------------------------------------------------------------
// Kernel 2: convert f32 weights -> bf16 (row-major [384][128])
// ---------------------------------------------------------------------------
__global__ __launch_bounds__(256) void tgn_cvt_kernel(const float* __restrict__ w,
                                                      __bf16* __restrict__ o,
                                                      int n) {
    int i = blockIdx.x * 256 + threadIdx.x;
    if (i < n) o[i] = (__bf16)w[i];
}

// ---------------------------------------------------------------------------
// Kernel 3: edge scatter — one wave32 per message (coalesced 512B per gather)
// msg = 2*e + dir : dir 0 -> src receives dst's features, dir 1 -> reverse
// ---------------------------------------------------------------------------
__global__ __launch_bounds__(256) void tgn_scatter_kernel(
    const int* __restrict__ src, const int* __restrict__ dst,
    const float* __restrict__ feat,
    float* __restrict__ aggSum, float* __restrict__ cnt, int numEdges) {

    const int msg  = (blockIdx.x * 256 + threadIdx.x) >> 5;
    const int lane = threadIdx.x & 31;
    if (msg >= 2 * numEdges) return;

    const int e   = msg >> 1;
    const int dir = msg & 1;
    const int s = src[e], d = dst[e];
    const int recv = dir ? d : s;
    const int nb   = dir ? s : d;

    const float4 f = ((const float4*)(feat + (size_t)nb * DIM))[lane];
    float* o = aggSum + (size_t)recv * DIM + lane * 4;
    atomicAdd(o + 0, f.x);
    atomicAdd(o + 1, f.y);
    atomicAdd(o + 2, f.z);
    atomicAdd(o + 3, f.w);
    if (lane == 0) atomicAdd(&cnt[recv], 1.0f);
}

// ---------------------------------------------------------------------------
// Kernel 4: fused GRU via bf16 WMMA.
// Workgroup = 16 nodes, 256 threads (8 waves). A fragments (agg and memory,
// 16x128 bf16 each) are loaded from LDS ONCE per wave (they are N-tile
// invariant), then each wave computes 3 N-tiles of both gi = agg @ w_ih^T and
// gh = h @ w_hh^T (16x16 f32 tiles, K=128 in four 16x16x32 bf16 WMMA steps).
// Results staged in LDS; fused elementwise gate epilogue follows.
// ---------------------------------------------------------------------------
__global__ __launch_bounds__(256) void tgn_gru_wmma_kernel(
    const float* __restrict__ aggSum, const float* __restrict__ cnt,
    const float* __restrict__ memory,
    const __bf16* __restrict__ wihB, const __bf16* __restrict__ whhB,
    const float* __restrict__ b_ih, const float* __restrict__ b_hh,
    float* __restrict__ out) {

    __shared__ __bf16 sAx[MT][DIM];    // mean-aggregated features, bf16
    __shared__ __bf16 sAh[MT][DIM];    // memory (hidden), bf16
    __shared__ float  sGi[MT][GDIM];   // gi accumulators (16 x 384)
    __shared__ float  sGh[MT][GDIM];   // gh accumulators
    __shared__ float  sCnt[MT];

    const int tid      = threadIdx.x;
    const int nodeBase = blockIdx.x * MT;

    // ---- Phase 0: load, mean-normalize, convert to bf16 in LDS ----
    if (tid < MT) sCnt[tid] = cnt[nodeBase + tid];
    for (int i = tid; i < MT * DIM; i += 256) {
        const int m = i >> 7, d = i & (DIM - 1);
        const float c   = cnt[nodeBase + m];
        const float inv = 1.0f / fmaxf(c, 1.0f);
        const float a = aggSum[(size_t)(nodeBase + m) * DIM + d] * inv;
        const float h = memory[(size_t)(nodeBase + m) * DIM + d];
        sAx[m][d] = (__bf16)a;
        sAh[m][d] = (__bf16)h;
    }
    __syncthreads();

    // ---- Phase 1: WMMA tiles ----
    const int wave = tid >> 5;
    const int lane = tid & 31;
    const int row  = lane & 15;   // A row / B col / C col
    const int half = lane >> 4;

    union frag { v16bf v; v8bf h[2]; };

    // Hoisted A fragments: tile-invariant, loaded once per wave.
    // A frag layout (16x32 bf16): VGPR0..3 K = kb+8*half..+7,
    //                             VGPR4..7 K = kb+16+8*half..+7
    frag ax[4], ah[4];
#pragma unroll
    for (int ks = 0; ks < 4; ++ks) {
        const int kb  = ks * 32;
        const int klo = kb + 8 * half;
        const int khi = kb + 16 + 8 * half;
        ax[ks].h[0] = *(const v8bf*)&sAx[row][klo];
        ax[ks].h[1] = *(const v8bf*)&sAx[row][khi];
        ah[ks].h[0] = *(const v8bf*)&sAh[row][klo];
        ah[ks].h[1] = *(const v8bf*)&sAh[row][khi];
    }

#pragma unroll
    for (int t = 0; t < 3; ++t) {
        const int tile  = wave * 3 + t;   // 0..23
        const int jbase = tile * 16;      // gate-output column base
        v8f accI = {};
        v8f accH = {};
#pragma unroll
        for (int ks = 0; ks < 4; ++ks) {
            const int kb = ks * 32;
            frag bi, bh;
            // B frag (32x16 bf16): col n = row, K = kb+8*half..+7, kb+16+8*half..+7
            // B[k][n] = W[jbase+n][k]  (computing x @ W^T)
            const int j  = jbase + row;
            const int kw = kb + 8 * half;
            bi.h[0] = *(const v8bf*)&wihB[(size_t)j * DIM + kw];
            bi.h[1] = *(const v8bf*)&wihB[(size_t)j * DIM + kw + 16];
            bh.h[0] = *(const v8bf*)&whhB[(size_t)j * DIM + kw];
            bh.h[1] = *(const v8bf*)&whhB[(size_t)j * DIM + kw + 16];

            accI = __builtin_amdgcn_wmma_f32_16x16x32_bf16(
                false, ax[ks].v, false, bi.v, (short)0, accI, false, false);
            accH = __builtin_amdgcn_wmma_f32_16x16x32_bf16(
                false, ah[ks].v, false, bh.v, (short)0, accH, false, false);
        }
        // C/D layout: lane col n=row, rows v + 8*half
#pragma unroll
        for (int v = 0; v < 8; ++v) {
            sGi[v + 8 * half][jbase + row] = accI[v];
            sGh[v + 8 * half][jbase + row] = accH[v];
        }
    }
    __syncthreads();

    // ---- Phase 2: elementwise GRU gates ----
    for (int i = tid; i < MT * DIM; i += 256) {
        const int m = i >> 7, d = i & (DIM - 1);
        const float ir = sGi[m][d]           + b_ih[d];
        const float iz = sGi[m][DIM + d]     + b_ih[DIM + d];
        const float in = sGi[m][2 * DIM + d] + b_ih[2 * DIM + d];
        const float hr = sGh[m][d]           + b_hh[d];
        const float hz = sGh[m][DIM + d]     + b_hh[DIM + d];
        const float hn = sGh[m][2 * DIM + d] + b_hh[2 * DIM + d];

        const float r  = 1.0f / (1.0f + __expf(-(ir + hr)));
        const float z  = 1.0f / (1.0f + __expf(-(iz + hz)));
        const float n  = tanhf(in + r * hn);
        const float h  = memory[(size_t)(nodeBase + m) * DIM + d];
        const float up = (1.0f - z) * n + z * h;
        out[(size_t)(nodeBase + m) * DIM + d] = (sCnt[m] > 0.0f) ? up : h;
    }
}

// ---------------------------------------------------------------------------
extern "C" void kernel_launch(void* const* d_in, const int* in_sizes, int n_in,
                              void* d_out, int out_size, void* d_ws, size_t ws_size,
                              hipStream_t stream) {
    const int*   src   = (const int*)  d_in[0];
    const int*   dst   = (const int*)  d_in[1];
    // d_in[2] = t (unused scalar)
    const float* feat  = (const float*)d_in[3];
    const float* mem   = (const float*)d_in[4];
    const float* w_ih  = (const float*)d_in[5];
    const float* w_hh  = (const float*)d_in[6];
    const float* b_ih  = (const float*)d_in[7];
    const float* b_hh  = (const float*)d_in[8];
    float*       out   = (float*)d_out;

    const int E = in_sizes[0];
    const int N = in_sizes[4] / DIM;       // 200000
    const int W = GDIM * DIM;              // 49152 weights per matrix

    // workspace layout: aggSum[N*DIM] f32 | cnt[N] f32 | wihB[W] bf16 | whhB[W] bf16
    float*  aggSum = (float*)d_ws;
    float*  cntBuf = aggSum + (size_t)N * DIM;
    __bf16* wihB   = (__bf16*)(cntBuf + N);
    __bf16* whhB   = wihB + W;

    // 1) zero aggregation buffers
    {
        const int total = N * DIM + N;
        tgn_zero_kernel<<<(total + 255) / 256, 256, 0, stream>>>(aggSum, total);
    }
    // 2) weights -> bf16
    tgn_cvt_kernel<<<(W + 255) / 256, 256, 0, stream>>>(w_ih, wihB, W);
    tgn_cvt_kernel<<<(W + 255) / 256, 256, 0, stream>>>(w_hh, whhB, W);
    // 3) edge scatter (one wave per message, 2E messages, 8 messages/block)
    {
        const int msgs   = 2 * E;
        const int blocks = (msgs * 32 + 255) / 256;
        tgn_scatter_kernel<<<blocks, 256, 0, stream>>>(src, dst, feat,
                                                       aggSum, cntBuf, E);
    }
    // 4) fused GRU with WMMA (16 nodes per block)
    {
        const int blocks = (N + MT - 1) / MT;   // 12500
        tgn_gru_wmma_kernel<<<blocks, 256, 0, stream>>>(aggSum, cntBuf, mem,
                                                        wihB, whhB, b_ih, b_hh,
                                                        out);
    }
}
